// DiffusionHybridMoEBlock_51883204935744
// MI455X (gfx1250) — compile-verified, matrix operand order checked by reading, not run on hardware
//
#include <hip/hip_runtime.h>
#include <hip/hip_bf16.h>
#include <math.h>
#include <stdint.h>

// ---------------------------------------------------------------------------
// Types for gfx1250 WMMA (wave32): v_wmma_f32_16x16x32_bf16
// ---------------------------------------------------------------------------
typedef __attribute__((ext_vector_type(16))) __bf16 v16bf;
typedef __attribute__((ext_vector_type(8)))  float  v8f;

union Frag { uint4 u[2]; v16bf v; };

#define TOKENS 4096   // B*T
#define DMODEL 1024
#define NHEAD  8
#define DHEAD  128
#define NEXP   8
#define DFF    2048
#define SEQT   1024
#define NBATCH 4

__device__ __forceinline__ uint16_t f2bf(float x) {
  uint32_t u = __float_as_uint(x);
  u += 0x7fffu + ((u >> 16) & 1u);            // round-to-nearest-even
  return (uint16_t)(u >> 16);
}
__device__ __forceinline__ float bf2f(uint16_t h) {
  return __uint_as_float(((uint32_t)h) << 16);
}

// CDNA5 async global->LDS copy (ASYNCcnt path), 16B per lane, no VGPR staging.
// LDS aperture: generic addr low 32 bits == LDS byte address.
__device__ __forceinline__ void async_copy_b128(const void* g, void* l) {
  uint32_t loff = (uint32_t)(uintptr_t)l;
  asm volatile("global_load_async_to_lds_b128 %0, %1, off"
               :: "v"(loff), "v"(g) : "memory");
}
__device__ __forceinline__ void wait_async0() {
  asm volatile("s_wait_asynccnt 0x0" ::: "memory");
}

// ---------------------------------------------------------------------------
// Epilogue functors
// ---------------------------------------------------------------------------
struct EpiQf {                       // elu(x)+1, token-major bf16 [4096,1024]
  uint16_t* C;
  __device__ void operator()(int, int m, int n, float v) const {
    float r = v > 0.f ? v + 1.f : __expf(v);
    C[(size_t)m * DMODEL + n] = f2bf(r);
  }
};
struct EpiBHDT {                     // store to [B,H,DH,T] bf16, optional elu+1
  uint16_t* C; int act;
  __device__ void operator()(int, int m, int n, float v) const {
    if (act) v = v > 0.f ? v + 1.f : __expf(v);
    int b = m >> 10, t = m & 1023, h = n >> 7, d = n & 127;
    size_t idx = (((size_t)(b * NHEAD + h) * DHEAD + d) << 10) + t;
    C[idx] = f2bf(v);
  }
};
struct EpiKvT {                      // store transposed kv^T [bh][e][d] bf16
  uint16_t* C;
  __device__ void operator()(int bz, int m, int n, float v) const {
    C[(size_t)bz * (DHEAD * DHEAD) + (size_t)n * DHEAD + m] = f2bf(v);
  }
};
struct EpiAttn {                     // divide by denom, store token-major bf16
  uint16_t* C; const float* denom;
  __device__ void operator()(int bz, int m, int n, float v) const {
    float d = denom[(size_t)bz * SEQT + m];
    int b = bz >> 3, h = bz & 7;
    size_t idx = ((size_t)b * SEQT + m) * DMODEL + h * DHEAD + n;
    C[idx] = f2bf(v / d);
  }
};
struct EpiQRes {                     // q + attn + bo -> fp32 q_res
  float* C; const float* q; const float* bo;
  __device__ void operator()(int, int m, int n, float v) const {
    size_t idx = (size_t)m * DMODEL + n;
    C[idx] = q[idx] + v + bo[n];
  }
};
struct EpiGelu {                     // exact gelu(x + b1) -> bf16 h [4096,2048]
  uint16_t* C; const float* bias;
  __device__ void operator()(int, int m, int n, float v) const {
    float x = v + bias[n];
    float g = 0.5f * x * (1.f + erff(x * 0.70710678118654752f));
    C[(size_t)m * DFF + n] = f2bf(g);
  }
};
struct EpiMoE {                      // moe += combine[m,e] * (x + b2)
  float* moe; const float* bias; const float* combine; int e;
  __device__ void operator()(int, int m, int n, float v) const {
    float w = combine[(size_t)m * NEXP + e];
    moe[(size_t)m * DMODEL + n] += w * (v + bias[n]);
  }
};

// ---------------------------------------------------------------------------
// Block-tile bf16 GEMM: C[M,N] = A[M,K] * B^T  (B stored as Bt[N,K])
// Block tile 128x128, 8 waves (4x2), wave tile 32x64 (2x4 WMMA accumulators).
// Double-buffered LDS fed by global_load_async_to_lds_b128 (ASYNCcnt),
// one barrier per K step. Batched via blockIdx.z (z>>3, z&7) strides.
// ---------------------------------------------------------------------------
#define LPITCH 40            // 32 + 8 bf16 pad -> 80B pitch, conflict-free b128
#define LTILE  (128 * LPITCH)

template <class Epi>
__global__ __launch_bounds__(256) void gemm_bf16(
    const uint16_t* __restrict__ A, int lda, long long sA0, long long sA1,
    const uint16_t* __restrict__ B, int ldb, long long sB0, long long sB1,
    int K, Epi epi)
{
  __shared__ uint16_t sAm[2 * LTILE];
  __shared__ uint16_t sBm[2 * LTILE];
  const int tid = threadIdx.x;
  const int bz  = blockIdx.z;
  const int m0  = blockIdx.x * 128;
  const int n0  = blockIdx.y * 128;
  const uint16_t* Ab = A + (size_t)(bz >> 3) * sA1 + (size_t)(bz & 7) * sA0;
  const uint16_t* Bb = B + (size_t)(bz >> 3) * sB1 + (size_t)(bz & 7) * sB0;

  const int wave = tid >> 5, lane = tid & 31;
  const int wm = wave >> 1, wn = wave & 1;
  const int lr = lane & 15, hi = lane >> 4;
  const int kbase = hi ? 8 : 0;  // ISA 16-bit A layout: lanes>=16 hold K 8-15/24-31

  // this thread's chunk coordinates (16B chunks; 4 per 64B row)
  const int row0 = tid >> 2,          kc0 = (tid & 3) * 8;
  const int row1 = (tid + 256) >> 2,  kc1 = kc0;  // (c&3) identical for +256

  v8f zero = {0.f, 0.f, 0.f, 0.f, 0.f, 0.f, 0.f, 0.f};
  v8f acc[2][4];
#pragma unroll
  for (int i = 0; i < 2; ++i)
#pragma unroll
    for (int j = 0; j < 4; ++j) acc[i][j] = zero;

  auto issue = [&](int bi, int k0) {
    async_copy_b128(Ab + (size_t)(m0 + row0) * lda + k0 + kc0,
                    &sAm[bi * LTILE + row0 * LPITCH + kc0]);
    async_copy_b128(Bb + (size_t)(n0 + row0) * ldb + k0 + kc0,
                    &sBm[bi * LTILE + row0 * LPITCH + kc0]);
    async_copy_b128(Ab + (size_t)(m0 + row1) * lda + k0 + kc1,
                    &sAm[bi * LTILE + row1 * LPITCH + kc1]);
    async_copy_b128(Bb + (size_t)(n0 + row1) * ldb + k0 + kc1,
                    &sBm[bi * LTILE + row1 * LPITCH + kc1]);
  };

  const int nk = K >> 5;
  issue(0, 0);
  for (int k = 0; k < nk; ++k) {
    const int bi = k & 1;
    wait_async0();        // this wave's tile-k copies landed in LDS
    __syncthreads();      // all waves' copies landed; prev compute finished
    if (k + 1 < nk) issue(1 - bi, (k + 1) << 5);  // overlap next tile

    const uint16_t* la = &sAm[bi * LTILE];
    const uint16_t* lb = &sBm[bi * LTILE];
    Frag fa[2], fb[4];
#pragma unroll
    for (int tm = 0; tm < 2; ++tm) {
      int r = wm * 32 + tm * 16 + lr;
      fa[tm].u[0] = *(const uint4*)(&la[r * LPITCH + kbase]);
      fa[tm].u[1] = *(const uint4*)(&la[r * LPITCH + kbase + 16]);
    }
#pragma unroll
    for (int tn = 0; tn < 4; ++tn) {
      int rc = wn * 64 + tn * 16 + lr;
      fb[tn].u[0] = *(const uint4*)(&lb[rc * LPITCH + kbase]);
      fb[tn].u[1] = *(const uint4*)(&lb[rc * LPITCH + kbase + 16]);
    }
#pragma unroll
    for (int tm = 0; tm < 2; ++tm)
#pragma unroll
      for (int tn = 0; tn < 4; ++tn)
        acc[tm][tn] = __builtin_amdgcn_wmma_f32_16x16x32_bf16(
            false, fa[tm].v, false, fb[tn].v, (short)0, acc[tm][tn],
            false, false);
  }

  // Epilogue: C/D layout — VGPR r: lanes 0-15 -> M=r, lanes 16-31 -> M=r+8
#pragma unroll
  for (int tm = 0; tm < 2; ++tm)
#pragma unroll
    for (int tn = 0; tn < 4; ++tn)
#pragma unroll
      for (int r = 0; r < 8; ++r) {
        int m = m0 + wm * 32 + tm * 16 + (hi ? r + 8 : r);
        int n = n0 + wn * 64 + tn * 16 + lr;
        epi(bz, m, n, acc[tm][tn][r]);
      }
}

// ---------------------------------------------------------------------------
// Weight convert + transpose: fp32 [R,C] -> bf16 [C,R]
// ---------------------------------------------------------------------------
__global__ __launch_bounds__(256) void transpose_f32_bf16(
    const float* __restrict__ in, uint16_t* __restrict__ out, int R, int C)
{
  __shared__ float t[32][33];
  int tx = threadIdx.x & 31, ty = threadIdx.x >> 5;
  int c0 = blockIdx.x * 32, r0 = blockIdx.y * 32;
#pragma unroll
  for (int j = 0; j < 4; ++j)
    t[ty + j * 8][tx] = in[(size_t)(r0 + ty + j * 8) * C + c0 + tx];
  __syncthreads();
#pragma unroll
  for (int j = 0; j < 4; ++j) {
    int oc = ty + j * 8;
    out[(size_t)(c0 + oc) * R + r0 + tx] = f2bf(t[tx][oc]);
  }
}

// ---------------------------------------------------------------------------
// LayerNorm over D=1024, one block per row; optional bf16 + fp32 outputs
// ---------------------------------------------------------------------------
__global__ __launch_bounds__(256) void ln_kernel(
    const float* __restrict__ x, const float* __restrict__ g,
    const float* __restrict__ b, uint16_t* __restrict__ obf,
    float* __restrict__ of)
{
  __shared__ float red[256];
  const int tid = threadIdx.x;
  size_t row = blockIdx.x;
  const float* xr = x + row * DMODEL;
  float v[4]; float s = 0.f;
#pragma unroll
  for (int i = 0; i < 4; ++i) { v[i] = xr[tid + 256 * i]; s += v[i]; }
  red[tid] = s; __syncthreads();
  for (int o = 128; o > 0; o >>= 1) { if (tid < o) red[tid] += red[tid + o]; __syncthreads(); }
  float mu = red[0] * (1.f / DMODEL);
  __syncthreads();
  float s2 = 0.f;
#pragma unroll
  for (int i = 0; i < 4; ++i) { v[i] -= mu; s2 += v[i] * v[i]; }
  red[tid] = s2; __syncthreads();
  for (int o = 128; o > 0; o >>= 1) { if (tid < o) red[tid] += red[tid + o]; __syncthreads(); }
  float rs = rsqrtf(red[0] * (1.f / DMODEL) + 1e-5f);
#pragma unroll
  for (int i = 0; i < 4; ++i) {
    int c = tid + 256 * i;
    float y = v[i] * rs * g[c] + b[c];
    if (obf) obf[row * DMODEL + c] = f2bf(y);
    if (of)  of[row * DMODEL + c] = y;
  }
}

// Final: out = q_res + LN(moe + x)
__global__ __launch_bounds__(256) void final_kernel(
    const float* __restrict__ moe, const float* __restrict__ x,
    const float* __restrict__ g, const float* __restrict__ b,
    const float* __restrict__ qres, float* __restrict__ out)
{
  __shared__ float red[256];
  const int tid = threadIdx.x;
  size_t row = blockIdx.x;
  float v[4]; float s = 0.f;
#pragma unroll
  for (int i = 0; i < 4; ++i) {
    size_t idx = row * DMODEL + tid + 256 * i;
    v[i] = moe[idx] + x[idx]; s += v[i];
  }
  red[tid] = s; __syncthreads();
  for (int o = 128; o > 0; o >>= 1) { if (tid < o) red[tid] += red[tid + o]; __syncthreads(); }
  float mu = red[0] * (1.f / DMODEL);
  __syncthreads();
  float s2 = 0.f;
#pragma unroll
  for (int i = 0; i < 4; ++i) { v[i] -= mu; s2 += v[i] * v[i]; }
  red[tid] = s2; __syncthreads();
  for (int o = 128; o > 0; o >>= 1) { if (tid < o) red[tid] += red[tid + o]; __syncthreads(); }
  float rs = rsqrtf(red[0] * (1.f / DMODEL) + 1e-5f);
#pragma unroll
  for (int i = 0; i < 4; ++i) {
    int c = tid + 256 * i;
    size_t idx = row * DMODEL + c;
    out[idx] = qres[idx] + v[i] * rs * g[c] + b[c];
  }
}

// k_sum[b,h,d] = sum_t kf + 1e-6 ; one wave32 per row of kfT [4096,1024]
__global__ __launch_bounds__(256) void ksum_kernel(
    const uint16_t* __restrict__ kfT, float* __restrict__ ksum)
{
  int wave = threadIdx.x >> 5, lane = threadIdx.x & 31;
  int row = blockIdx.x * 8 + wave;
  const uint16_t* p = kfT + (size_t)row * SEQT;
  float s = 0.f;
  for (int i = 0; i < SEQT / 32; ++i) s += bf2f(p[lane + i * 32]);
  for (int o = 16; o > 0; o >>= 1) s += __shfl_xor(s, o, 32);
  if (lane == 0) ksum[row] = s + 1e-6f;
}

// denom[bh,t] = dot(qf[b,t,h,:], ksum[bh,:]) over DH=128; one wave per (bh,t)
__global__ __launch_bounds__(256) void denom_kernel(
    const uint16_t* __restrict__ qf, const float* __restrict__ ksum,
    float* __restrict__ denom)
{
  int wave = threadIdx.x >> 5, lane = threadIdx.x & 31;
  int idx = blockIdx.x * 8 + wave;           // 0 .. 32767
  int z = idx >> 10, t = idx & 1023;
  int b = z >> 3, h = z & 7;
  const uint16_t* qp = qf + ((size_t)b * SEQT + t) * DMODEL + h * DHEAD;
  const float* ks = ksum + (size_t)z * DHEAD;
  float s = 0.f;
#pragma unroll
  for (int i = 0; i < 4; ++i) { int d = lane + i * 32; s += bf2f(qp[d]) * ks[d]; }
  for (int o = 16; o > 0; o >>= 1) s += __shfl_xor(s, o, 32);
  if (lane == 0) denom[idx] = s;
}

// Noisy top-2 router: softmax(x@Wr + br + 0.1*noise), keep top-2 scores
__global__ __launch_bounds__(256) void router_kernel(
    const float* __restrict__ x, const float* __restrict__ Wr,
    const float* __restrict__ br, const float* __restrict__ noise,
    float* __restrict__ combine)
{
  int wave = threadIdx.x >> 5, lane = threadIdx.x & 31;
  size_t t = (size_t)blockIdx.x * 8 + wave;
  const float* xr = x + t * DMODEL;
  float p[8] = {0.f, 0.f, 0.f, 0.f, 0.f, 0.f, 0.f, 0.f};
  for (int i = 0; i < DMODEL / 32; ++i) {
    int d = lane + i * 32;
    float xv = xr[d];
    const float* w = Wr + (size_t)d * NEXP;
#pragma unroll
    for (int e = 0; e < 8; ++e) p[e] += xv * w[e];
  }
#pragma unroll
  for (int e = 0; e < 8; ++e)
    for (int o = 16; o > 0; o >>= 1) p[e] += __shfl_xor(p[e], o, 32);
  if (lane == 0) {
    float lg[8]; float mx = -1e30f;
#pragma unroll
    for (int e = 0; e < 8; ++e) {
      lg[e] = p[e] + br[e] + 0.1f * noise[t * 8 + e];
      mx = fmaxf(mx, lg[e]);
    }
    float se = 0.f;
#pragma unroll
    for (int e = 0; e < 8; ++e) { lg[e] = __expf(lg[e] - mx); se += lg[e]; }
    float inv = 1.f / se;
#pragma unroll
    for (int e = 0; e < 8; ++e) lg[e] *= inv;
    int i1 = 0;
#pragma unroll
    for (int e = 1; e < 8; ++e) if (lg[e] > lg[i1]) i1 = e;
    int i2 = (i1 == 0) ? 1 : 0;
#pragma unroll
    for (int e = 0; e < 8; ++e) if (e != i1 && lg[e] > lg[i2]) i2 = e;
#pragma unroll
    for (int e = 0; e < 8; ++e)
      combine[t * 8 + e] = (e == i1 || e == i2) ? lg[e] : 0.f;
  }
}

// ---------------------------------------------------------------------------
extern "C" void kernel_launch(void* const* d_in, const int* in_sizes, int n_in,
                              void* d_out, int out_size, void* d_ws, size_t ws_size,
                              hipStream_t stream)
{
  (void)in_sizes; (void)n_in; (void)out_size; (void)ws_size;
  const float* v_in  = (const float*)d_in[0];
  const float* k_in  = (const float*)d_in[1];
  const float* q_in  = (const float*)d_in[2];
  const float* noise = (const float*)d_in[3];
  const float* g_v = (const float*)d_in[4],  * b_v = (const float*)d_in[5];
  const float* g_k = (const float*)d_in[6],  * b_k = (const float*)d_in[7];
  const float* g_q = (const float*)d_in[8],  * b_q = (const float*)d_in[9];
  const float* g_m = (const float*)d_in[10], * b_m = (const float*)d_in[11];
  const float* g_o = (const float*)d_in[12], * b_o = (const float*)d_in[13];
  const float* Wq = (const float*)d_in[14];
  const float* Wk = (const float*)d_in[15];
  const float* Wv = (const float*)d_in[16];
  const float* Wo = (const float*)d_in[17];
  const float* bo = (const float*)d_in[18];
  const float* Wr = (const float*)d_in[19];
  const float* br = (const float*)d_in[20];
  const float* W1 = (const float*)d_in[21];
  const float* b1 = (const float*)d_in[22];
  const float* W2 = (const float*)d_in[23];
  const float* b2 = (const float*)d_in[24];

  char* p = (char*)d_ws;
  auto alloc = [&](size_t bytes) -> char* {
    char* r = p; p += (bytes + 255) & ~(size_t)255; return r;
  };
  const size_t NT = (size_t)TOKENS;
  uint16_t* WqT = (uint16_t*)alloc((size_t)DMODEL * DMODEL * 2);
  uint16_t* WkT = (uint16_t*)alloc((size_t)DMODEL * DMODEL * 2);
  uint16_t* WvT = (uint16_t*)alloc((size_t)DMODEL * DMODEL * 2);
  uint16_t* WoT = (uint16_t*)alloc((size_t)DMODEL * DMODEL * 2);
  uint16_t* W1T = (uint16_t*)alloc((size_t)NEXP * DFF * DMODEL * 2);
  uint16_t* W2T = (uint16_t*)alloc((size_t)NEXP * DMODEL * DFF * 2);
  uint16_t* qn  = (uint16_t*)alloc(NT * DMODEL * 2);
  uint16_t* kn  = (uint16_t*)alloc(NT * DMODEL * 2);
  uint16_t* vn  = (uint16_t*)alloc(NT * DMODEL * 2);
  uint16_t* qf  = (uint16_t*)alloc(NT * DMODEL * 2);
  uint16_t* kfT = (uint16_t*)alloc(NT * DMODEL * 2);
  uint16_t* vhT = (uint16_t*)alloc(NT * DMODEL * 2);
  uint16_t* kvT = (uint16_t*)alloc((size_t)NBATCH * NHEAD * DHEAD * DHEAD * 2);
  float* ksum   = (float*)alloc((size_t)NBATCH * NHEAD * DHEAD * 4);
  float* denom  = (float*)alloc((size_t)NBATCH * NHEAD * SEQT * 4);
  uint16_t* att = (uint16_t*)alloc(NT * DMODEL * 2);
  float* qres   = (float*)alloc(NT * DMODEL * 4);
  float* xbuf   = (float*)alloc(NT * DMODEL * 4);
  uint16_t* xf  = (uint16_t*)alloc(NT * DMODEL * 2);
  float* comb   = (float*)alloc(NT * NEXP * 4);
  uint16_t* hbf = (uint16_t*)alloc(NT * DFF * 2);
  float* moe    = (float*)alloc(NT * DMODEL * 4);

  dim3 blk(256);

  // 1) weight convert+transpose to bf16
  transpose_f32_bf16<<<dim3(32, 32), blk, 0, stream>>>(Wq, WqT, DMODEL, DMODEL);
  transpose_f32_bf16<<<dim3(32, 32), blk, 0, stream>>>(Wk, WkT, DMODEL, DMODEL);
  transpose_f32_bf16<<<dim3(32, 32), blk, 0, stream>>>(Wv, WvT, DMODEL, DMODEL);
  transpose_f32_bf16<<<dim3(32, 32), blk, 0, stream>>>(Wo, WoT, DMODEL, DMODEL);
  for (int e = 0; e < NEXP; ++e) {
    transpose_f32_bf16<<<dim3(64, 32), blk, 0, stream>>>(
        W1 + (size_t)e * DMODEL * DFF, W1T + (size_t)e * DFF * DMODEL, DMODEL, DFF);
    transpose_f32_bf16<<<dim3(32, 64), blk, 0, stream>>>(
        W2 + (size_t)e * DFF * DMODEL, W2T + (size_t)e * DMODEL * DFF, DFF, DMODEL);
  }

  // 2) pre-attention LayerNorms -> bf16
  ln_kernel<<<TOKENS, blk, 0, stream>>>(q_in, g_q, b_q, qn, nullptr);
  ln_kernel<<<TOKENS, blk, 0, stream>>>(k_in, g_k, b_k, kn, nullptr);
  ln_kernel<<<TOKENS, blk, 0, stream>>>(v_in, g_v, b_v, vn, nullptr);

  // 3) QKV projections (WMMA) with feature-map epilogues
  gemm_bf16<<<dim3(32, 8, 1), blk, 0, stream>>>(
      qn, DMODEL, 0, 0, WqT, DMODEL, 0, 0, DMODEL, EpiQf{qf});
  gemm_bf16<<<dim3(32, 8, 1), blk, 0, stream>>>(
      kn, DMODEL, 0, 0, WkT, DMODEL, 0, 0, DMODEL, EpiBHDT{kfT, 1});
  gemm_bf16<<<dim3(32, 8, 1), blk, 0, stream>>>(
      vn, DMODEL, 0, 0, WvT, DMODEL, 0, 0, DMODEL, EpiBHDT{vhT, 0});

  // 4) k_sum, kv = kf^T @ vh (batched over 32 b*h), denom, out = qf @ kv
  ksum_kernel<<<512, blk, 0, stream>>>(kfT, ksum);
  gemm_bf16<<<dim3(1, 1, 32), blk, 0, stream>>>(
      kfT, SEQT, 131072LL, 1048576LL,
      vhT, SEQT, 131072LL, 1048576LL, SEQT, EpiKvT{kvT});
  denom_kernel<<<4096, blk, 0, stream>>>(qf, ksum, denom);
  gemm_bf16<<<dim3(8, 1, 32), blk, 0, stream>>>(
      qf, DMODEL, 128LL, 1048576LL,
      kvT, DHEAD, 16384LL, 131072LL, DHEAD, EpiAttn{att, denom});

  // 5) out projection + bias + residual -> q_res (fp32)
  gemm_bf16<<<dim3(32, 8, 1), blk, 0, stream>>>(
      att, DMODEL, 0, 0, WoT, DMODEL, 0, 0, DMODEL, EpiQRes{qres, q_in, bo});

  // 6) MoE LayerNorm (x fp32 + xf bf16), router
  ln_kernel<<<TOKENS, blk, 0, stream>>>(qres, g_m, b_m, xf, xbuf);
  router_kernel<<<512, blk, 0, stream>>>(xbuf, Wr, br, noise, comb);

  // 7) dense experts (WMMA), combine-weighted accumulation into moe
  hipMemsetAsync(moe, 0, NT * DMODEL * 4, stream);
  for (int e = 0; e < NEXP; ++e) {
    gemm_bf16<<<dim3(32, 16, 1), blk, 0, stream>>>(
        xf, DMODEL, 0, 0, W1T + (size_t)e * DFF * DMODEL, DMODEL, 0, 0,
        DMODEL, EpiGelu{hbf, b1 + (size_t)e * DFF});
    gemm_bf16<<<dim3(32, 8, 1), blk, 0, stream>>>(
        hbf, DFF, 0, 0, W2T + (size_t)e * DMODEL * DFF, DFF, 0, 0,
        DFF, EpiMoE{moe, b2 + (size_t)e * DMODEL, comb, e});
  }

  // 8) out = q_res + LN(moe + x)
  final_kernel<<<TOKENS, blk, 0, stream>>>(moe, xbuf, g_o, b_o, qres,
                                           (float*)d_out);
}
